// CombineGraph_7275674600592
// MI455X (gfx1250) — matrix operand order repaired and checked by reading, not compiled, not run on hardware
//
#include <hip/hip_runtime.h>

typedef unsigned short u16;
typedef unsigned int   u32;
typedef __attribute__((ext_vector_type(8)))  float        v8f;
typedef __attribute__((ext_vector_type(8)))  unsigned int v8u;
typedef __attribute__((ext_vector_type(16))) __bf16       v16bf;

#define DEV __device__ __forceinline__

constexpr int   Bb = 128, Ll = 20, Dd = 256, Kk = 12;
constexpr int   BL = Bb * Ll;              // 2560
constexpr int   R1 = BL * Kk;              // 30720  (B * 240)
constexpr int   R2 = R1 * Kk;              // 368640 (B * 2880)
constexpr float SLOPE = 0.2f;
constexpr float BETAC = 0.005f;

// ---- bf16 helpers -----------------------------------------------------------
DEV u32 f2bf(float f) {
  u32 u = __builtin_bit_cast(u32, f);
  return (u + 0x7FFFu + ((u >> 16) & 1u)) >> 16;  // round-nearest-even
}

DEV v16bf pack_frag(uint4 a, uint4 b) {
  v8u r;
  r[0] = a.x; r[1] = a.y; r[2] = a.z; r[3] = a.w;
  r[4] = b.x; r[5] = b.y; r[6] = b.z; r[7] = b.w;
  return __builtin_bit_cast(v16bf, r);
}

// A fragment (16x32 bf16 tile staged in LDS, row-major, row stride 32 elems).
// lane l: m = l&15; elems e -> k = 16*(l>>4)+e
DEV v16bf load_a_frag_lds(const u16* ab) {
  int l  = threadIdx.x & 31;
  int m  = l & 15;
  int h8 = (l >> 4) * 16;
  const u16* row = ab + m * 32 + h8;
  uint4 lo = *(const uint4*)(row);
  uint4 hi = *(const uint4*)(row + 8);
  return pack_frag(lo, hi);
}

// B fragment (pre-swizzled in global: 32 lanes x 16 bf16 contiguous per lane)
DEV v16bf load_b_frag(const u16* wf) {
  int l = threadIdx.x & 31;
  const u16* p = wf + (size_t)l * 16;
  uint4 lo = *(const uint4*)(p);
  uint4 hi = *(const uint4*)(p + 8);
  return pack_frag(lo, hi);
}

// ---- generic 16-row x 256-col WMMA GEMM core --------------------------------
// 128 threads (4 waves). Wave w owns n-tiles [4w, 4w+4). Accum in f32.
// LDS stage is double-buffered (ab = 2 x 512 bf16) -> one barrier per k-chunk.
template <typename LoadA>
DEV void gemm_rows16(LoadA&& loadA, const u16* Wf, int nkt, u16* ab, v8f acc[4]) {
  const int tid  = threadIdx.x;
  const int wave = tid >> 5;
#pragma unroll
  for (int i = 0; i < 4; ++i)
#pragma unroll
    for (int j = 0; j < 8; ++j) acc[i][j] = 0.f;

  for (int kt = 0; kt < nkt; ++kt) {
    u16* buf   = ab + (kt & 1) * 512;
    u32* buf32 = (u32*)buf;
    // stage 16x32 f32 -> bf16, two packed dword stores per thread
#pragma unroll
    for (int q = 0; q < 2; ++q) {
      int p = q * 128 + tid;        // pair index 0..255
      int r = p >> 4;               // row 0..15
      int k = (p & 15) * 2;         // col 0,2,..,30
      u32 lo = f2bf(loadA(r, kt * 32 + k));
      u32 hi = f2bf(loadA(r, kt * 32 + k + 1));
      buf32[p] = lo | (hi << 16);
    }
    __syncthreads();
    v16bf afrag = load_a_frag_lds(buf);
    if (kt + 1 < nkt)  // keep next chunk's weight fragments warm (global_prefetch)
      __builtin_prefetch(Wf + (size_t)((kt + 1) * 16 + wave * 4) * 512, 0, 3);
    v16bf bf[4];
#pragma unroll
    for (int nt = 0; nt < 4; ++nt)
      bf[nt] = load_b_frag(Wf + (size_t)(kt * 16 + wave * 4 + nt) * 512);
#pragma unroll
    for (int nt = 0; nt < 4; ++nt)
      acc[nt] = __builtin_amdgcn_wmma_f32_16x16x32_bf16(
          false, afrag, false, bf[nt], (short)0, acc[nt], false, false);
  }
  __syncthreads();
}

// ---- weight swizzle: W[Kdim x 256] f32 -> WMMA B-fragments (bf16) -----------
__global__ void swizzle_w_kernel(const float* __restrict__ W, u16* __restrict__ Wf,
                                 int kdim) {
  int idx = blockIdx.x * 256 + threadIdx.x;
  int total = (kdim / 32) * 16 * 512;
  if (idx >= total) return;
  int e  = idx & 15;
  int l  = (idx >> 4) & 31;
  int nt = (idx >> 9) & 15;
  int kt = idx >> 13;
  int k  = kt * 32 + e + 16 * (l >> 4);
  int n  = nt * 16 + (l & 15);
  Wf[idx] = (u16)f2bf(W[(size_t)k * 256 + n]);
}

// ---- simple gathers / prep --------------------------------------------------
__global__ void gather_rows_kernel(const int* __restrict__ idx,
                                   const float* __restrict__ emb,
                                   float* __restrict__ out) {
  size_t row = blockIdx.x;
  int d = threadIdx.x;
  out[row * Dd + d] = emb[(size_t)idx[row] * Dd + d];
}

__global__ void nbr_kernel(const int* __restrict__ prev, const int* __restrict__ adj_all,
                           const float* __restrict__ num_w, int* __restrict__ nidx,
                           float* __restrict__ nw, int n) {
  int t = blockIdx.x * 256 + threadIdx.x;
  if (t >= n) return;
  int p  = prev[t / Kk];
  int kq = t % Kk;
  nidx[t] = adj_all[(size_t)p * Kk + kq];
  nw[t]   = num_w[(size_t)p * Kk + kq];
}

__global__ void sess_kernel(const int* __restrict__ item, const int* __restrict__ mask,
                            const float* __restrict__ emb, float* __restrict__ sess) {
  int b = blockIdx.x, d = threadIdx.x;
  float s = 0.f, cnt = 0.f;
  for (int l = 0; l < Ll; ++l) {
    float mf = (float)mask[b * Ll + l];
    cnt += mf;
    s += mf * emb[(size_t)item[b * Ll + l] * Dd + d];
  }
  sess[b * Dd + d] = s / cnt;
}

// ---- LocalAggregator: relation-typed GAT (STEP=1) ---------------------------
__global__ void local_agg_kernel(const float* __restrict__ h0, const int* __restrict__ adj,
                                 const float* __restrict__ la_a, float* __restrict__ hloc) {
  __shared__ float hbuf[Ll * Dd];
  __shared__ float att[Ll][Ll];
  int b = blockIdx.x, tid = threadIdx.x;
  for (int e = tid; e < Ll * Dd; e += 256) hbuf[e] = h0[(size_t)b * Ll * Dd + e];
  __syncthreads();
  for (int p = tid; p < Ll * Ll; p += 256) {
    int i = p / Ll, j = p % Ll;
    float a0 = 0.f, a1 = 0.f, a2 = 0.f, a3 = 0.f;
    for (int d = 0; d < Dd; ++d) {
      float x = hbuf[i * Dd + d] * hbuf[j * Dd + d];
      a0 += x * la_a[d * 4 + 0];
      a1 += x * la_a[d * 4 + 1];
      a2 += x * la_a[d * 4 + 2];
      a3 += x * la_a[d * 4 + 3];
    }
    int ad = adj[(size_t)b * Ll * Ll + p];
    float v = -9e15f;
    if (ad >= 1 && ad <= 4) {
      float t = (ad == 1) ? a0 : (ad == 2) ? a1 : (ad == 3) ? a2 : a3;
      v = (t >= 0.f) ? t : SLOPE * t;
    }
    att[i][j] = v;
  }
  __syncthreads();
  if (tid < Ll) {
    int i = tid;
    float mx = -3.4e38f;
    for (int j = 0; j < Ll; ++j) mx = fmaxf(mx, att[i][j]);
    float s = 0.f;
    for (int j = 0; j < Ll; ++j) { float e = __expf(att[i][j] - mx); att[i][j] = e; s += e; }
    float inv = 1.f / s;
    for (int j = 0; j < Ll; ++j) att[i][j] *= inv;
  }
  __syncthreads();
  for (int e = tid; e < Ll * Dd; e += 256) {
    int i = e / Dd, d = e % Dd;
    float s = 0.f;
    for (int j = 0; j < Ll; ++j) s += att[i][j] * hbuf[j * Dd + d];
    hloc[(size_t)b * Ll * Dd + e] = s;
  }
}

// ---- GEMM: gather rows of emb, plain store (tgt @ ta_w2) --------------------
__global__ void gemm_gather_store_kernel(const int* __restrict__ idx,
                                         const float* __restrict__ emb,
                                         const u16* __restrict__ Wf,
                                         float* __restrict__ out) {
  __shared__ __align__(16) u16 ab[1024];
  int mbase = blockIdx.x * 16;
  auto loadA = [&](int r, int k) -> float {
    return emb[(size_t)idx[mbase + r] * Dd + k];
  };
  v8f acc[4];
  gemm_rows16(loadA, Wf, 8, ab, acc);
  int l = threadIdx.x & 31, wave = threadIdx.x >> 5, lo = l & 15, hi = l >> 4;
#pragma unroll
  for (int nt = 0; nt < 4; ++nt) {
    int n = wave * 64 + nt * 16 + lo;
#pragma unroll
    for (int rr = 0; rr < 8; ++rr)
      out[(size_t)(mbase + rr + 8 * hi) * Dd + n] = acc[nt][rr];
  }
}

// ---- GEMM: target gate  h_target -> d_out[:, 0:20, :] -----------------------
__global__ void gemm_gate_kernel(const float* __restrict__ h0, const u16* __restrict__ Wf,
                                 const float* __restrict__ tw2, const float* __restrict__ tab,
                                 const int* __restrict__ mask, float* __restrict__ dout) {
  __shared__ __align__(16) u16 ab[1024];
  int mbase = blockIdx.x * 16;
  auto loadA = [&](int r, int k) -> float { return h0[(size_t)(mbase + r) * Dd + k]; };
  v8f acc[4];
  gemm_rows16(loadA, Wf, 8, ab, acc);
  int l = threadIdx.x & 31, wave = threadIdx.x >> 5, lo = l & 15, hi = l >> 4;
#pragma unroll
  for (int nt = 0; nt < 4; ++nt) {
    int n = wave * 64 + nt * 16 + lo;
#pragma unroll
    for (int rr = 0; rr < 8; ++rr) {
      int m = mbase + rr + 8 * hi;
      int b = m / Ll, li = m % Ll;
      float g = acc[nt][rr] + tw2[(size_t)b * Dd + n] + tab[n];
      g = 1.f / (1.f + __expf(-g));
      float mf = (float)mask[m];
      dout[((size_t)b * (2 * Ll) + li) * Dd + n] = g * h0[(size_t)m * Dd + n] * mf;
    }
  }
}

// ---- GEMM: attention logits alpha[m] = leaky(X@W + nw*w1last) . gw2 ---------
__global__ void gemm_alpha_kernel(const float* __restrict__ sess, const float* __restrict__ src,
                                  const int* __restrict__ idx, const float* __restrict__ emb,
                                  const float* __restrict__ nw, const u16* __restrict__ Wf,
                                  const float* __restrict__ w1last, const float* __restrict__ g2,
                                  int sk, float* __restrict__ alpha) {
  __shared__ __align__(16) u16 ab[1024];
  __shared__ float red[4][16];
  int mbase = blockIdx.x * 16;
  auto loadA = [&](int r, int k) -> float {
    int m = mbase + r;
    const float* rp = idx ? (emb + (size_t)idx[m] * Dd) : (src + (size_t)m * Dd);
    return sess[(size_t)(m / sk) * Dd + k] * rp[k];
  };
  v8f acc[4];
  gemm_rows16(loadA, Wf, 8, ab, acc);
  int tid = threadIdx.x, l = tid & 31, wave = tid >> 5, lo = l & 15, hi = l >> 4;
  float nwv[8];
#pragma unroll
  for (int rr = 0; rr < 8; ++rr) nwv[rr] = nw[mbase + rr + 8 * hi];
  float part[8] = {0.f, 0.f, 0.f, 0.f, 0.f, 0.f, 0.f, 0.f};
#pragma unroll
  for (int nt = 0; nt < 4; ++nt) {
    int n = wave * 64 + nt * 16 + lo;
    float wl = w1last[n], w2n = g2[n];
#pragma unroll
    for (int rr = 0; rr < 8; ++rr) {
      float v = acc[nt][rr] + nwv[rr] * wl;
      v = (v >= 0.f) ? v : SLOPE * v;
      part[rr] += v * w2n;
    }
  }
#pragma unroll
  for (int rr = 0; rr < 8; ++rr) {
#pragma unroll
    for (int s = 1; s < 16; s <<= 1) part[rr] += __shfl_xor(part[rr], s, 32);
  }
  if (lo == 0) {
#pragma unroll
    for (int rr = 0; rr < 8; ++rr) red[wave][hi * 8 + rr] = part[rr];
  }
  __syncthreads();
  if (tid < 16)
    alpha[mbase + tid] = red[0][tid] + red[1][tid] + red[2][tid] + red[3][tid];
}

// ---- softmax over K=12 + weighted neighbor sum ------------------------------
__global__ void softmax_aggn_kernel(const float* __restrict__ alpha,
                                    const float* __restrict__ nvsrc,
                                    const int* __restrict__ idx,
                                    const float* __restrict__ emb,
                                    float* __restrict__ aggn) {
  size_t row = blockIdx.x;
  int d = threadIdx.x;
  float a[Kk];
  float mx = -3.4e38f;
#pragma unroll
  for (int k = 0; k < Kk; ++k) { a[k] = alpha[row * Kk + k]; mx = fmaxf(mx, a[k]); }
  float s = 0.f;
#pragma unroll
  for (int k = 0; k < Kk; ++k) { a[k] = __expf(a[k] - mx); s += a[k]; }
  float inv = 1.f / s;
  float acc = 0.f;
#pragma unroll
  for (int k = 0; k < Kk; ++k) {
    size_t nr = row * Kk + k;
    const float* rp = idx ? (emb + (size_t)idx[nr] * Dd) : (nvsrc + nr * Dd);
    acc += a[k] * inv * rp[d];
  }
  aggn[row * Dd + d] = acc;
}

// ---- GEMM: relu(concat(sv, aggn) @ gw3), K = 512 ----------------------------
__global__ void gemm_relu_cat_kernel(const float* __restrict__ sv, const float* __restrict__ agg,
                                     const u16* __restrict__ Wf, float* __restrict__ dst) {
  __shared__ __align__(16) u16 ab[1024];
  int mbase = blockIdx.x * 16;
  auto loadA = [&](int r, int k) -> float {
    int m = mbase + r;
    return (k < Dd) ? sv[(size_t)m * Dd + k] : agg[(size_t)m * Dd + (k - Dd)];
  };
  v8f acc[4];
  gemm_rows16(loadA, Wf, 16, ab, acc);
  int l = threadIdx.x & 31, wave = threadIdx.x >> 5, lo = l & 15, hi = l >> 4;
#pragma unroll
  for (int nt = 0; nt < 4; ++nt) {
    int n = wave * 64 + nt * 16 + lo;
#pragma unroll
    for (int rr = 0; rr < 8; ++rr)
      dst[(size_t)(mbase + rr + 8 * hi) * Dd + n] = fmaxf(acc[nt][rr], 0.f);
  }
}

// ---- output second half: h_local + h_global ---------------------------------
__global__ void addout_kernel(const float* __restrict__ hl, const float* __restrict__ hg,
                              float* __restrict__ dout) {
  size_t t = (size_t)blockIdx.x * 256 + threadIdx.x;   // over B*L*D
  int d = (int)(t & 255);
  int m = (int)(t >> 8);
  int b = m / Ll, li = m % Ll;
  dout[((size_t)b * (2 * Ll) + Ll + li) * Dd + d] = hl[t] + hg[t];
}

// ---- SSL contrastive loss ---------------------------------------------------
__global__ void con_part_kernel(const float* __restrict__ hl, const float* __restrict__ hg,
                                float* __restrict__ part) {
  int b = blockIdx.x, d = threadIdx.x;
  int pb = (b * 37 + 11) & (Bb - 1);   // deterministic batch permutation
  float pos = 0.f, neg = 0.f;
  for (int li = 0; li < Ll; ++li) {
    int pl = (li * 7 + 3) % Ll;        // deterministic position permutation
    float g = hg[((size_t)b * Ll + li) * Dd + d];
    pos += hl[((size_t)b * Ll + li) * Dd + d] * g;
    neg += g * hl[((size_t)pb * Ll + pl) * Dd + d];
  }
  float sp = 1.f / (1.f + __expf(-pos));
  float sn = 1.f / (1.f + __expf(-neg));
  float val = -__logf(1e-8f + sp) - __logf(1e-8f + 1.f - sn);
  __shared__ float red[256];
  red[d] = val;
  __syncthreads();
  for (int s = 128; s > 0; s >>= 1) {
    if (d < s) red[d] += red[d + s];
    __syncthreads();
  }
  if (d == 0) part[b] = red[0];
}

__global__ void con_reduce_kernel(const float* __restrict__ part, float* __restrict__ dout,
                                  int outlast) {
  __shared__ float red[128];
  int t = threadIdx.x;
  red[t] = part[t];
  __syncthreads();
  for (int s = 64; s > 0; s >>= 1) {
    if (t < s) red[t] += red[t + s];
    __syncthreads();
  }
  if (t == 0) dout[outlast] = BETAC * red[0];
}

// ============================================================================
extern "C" void kernel_launch(void* const* d_in, const int* in_sizes, int n_in,
                              void* d_out, int out_size, void* d_ws, size_t ws_size,
                              hipStream_t stream) {
  (void)in_sizes; (void)n_in; (void)ws_size;
  const int*   inputs  = (const int*)d_in[0];
  const int*   adj     = (const int*)d_in[1];
  const int*   mask    = (const int*)d_in[2];
  const int*   item    = (const int*)d_in[3];
  const int*   targets = (const int*)d_in[4];
  const int*   adj_all = (const int*)d_in[5];
  const float* emb     = (const float*)d_in[6];
  const float* la_a    = (const float*)d_in[7];
  const float* ta_w1   = (const float*)d_in[8];
  const float* ta_w2   = (const float*)d_in[9];
  const float* ta_b    = (const float*)d_in[10];
  const float* gw1     = (const float*)d_in[11];
  const float* gw2     = (const float*)d_in[12];
  const float* gw3     = (const float*)d_in[13];
  const float* num_w   = (const float*)d_in[14];
  float* dout = (float*)d_out;

  // ---- workspace layout (256B aligned slots) ----
  char* w = (char*)d_ws;
  auto alloc = [&](size_t bytes) -> char* {
    char* p = w;
    w += (bytes + 255) & ~(size_t)255;
    return p;
  };
  float* h0     = (float*)alloc((size_t)BL * Dd * 4);
  float* hloc   = (float*)alloc((size_t)BL * Dd * 4);
  float* sess   = (float*)alloc((size_t)Bb * Dd * 4);
  float* tw2b   = (float*)alloc((size_t)Bb * Dd * 4);
  int*   n1i    = (int*)alloc((size_t)R1 * 4);
  float* w1v    = (float*)alloc((size_t)R1 * 4);
  int*   n2i    = (int*)alloc((size_t)R2 * 4);
  float* w2v    = (float*)alloc((size_t)R2 * 4);
  float* e1     = (float*)alloc((size_t)R1 * Dd * 4);
  float* aggb   = (float*)alloc((size_t)R1 * Dd * 4);
  float* ent1n  = (float*)alloc((size_t)R1 * Dd * 4);
  float* ent0n  = (float*)alloc((size_t)BL * Dd * 4);
  float* hglob  = (float*)alloc((size_t)BL * Dd * 4);
  float* alphab = (float*)alloc((size_t)R2 * 4);
  float* conp   = (float*)alloc((size_t)Bb * 4);
  u16* taw1f = (u16*)alloc((size_t)65536 * 2);
  u16* taw2f = (u16*)alloc((size_t)65536 * 2);
  u16* gw1f0 = (u16*)alloc((size_t)65536 * 2);
  u16* gw1f1 = (u16*)alloc((size_t)65536 * 2);
  u16* gw3f0 = (u16*)alloc((size_t)131072 * 2);
  u16* gw3f1 = (u16*)alloc((size_t)131072 * 2);

  // ---- weight pre-swizzle to WMMA fragments ----
  swizzle_w_kernel<<<256, 256, 0, stream>>>(ta_w1, taw1f, 256);
  swizzle_w_kernel<<<256, 256, 0, stream>>>(ta_w2, taw2f, 256);
  swizzle_w_kernel<<<256, 256, 0, stream>>>(gw1, gw1f0, 256);                  // hop0, rows 0..255
  swizzle_w_kernel<<<256, 256, 0, stream>>>(gw1 + 257 * 256, gw1f1, 256);      // hop1
  swizzle_w_kernel<<<512, 256, 0, stream>>>(gw3, gw3f0, 512);
  swizzle_w_kernel<<<512, 256, 0, stream>>>(gw3 + 512 * 256, gw3f1, 512);

  // ---- gathers / prep ----
  gather_rows_kernel<<<BL, 256, 0, stream>>>(inputs, emb, h0);
  sess_kernel<<<Bb, 256, 0, stream>>>(item, mask, emb, sess);
  nbr_kernel<<<(R1 + 255) / 256, 256, 0, stream>>>(inputs, adj_all, num_w, n1i, w1v, R1);
  nbr_kernel<<<(R2 + 255) / 256, 256, 0, stream>>>(n1i, adj_all, num_w, n2i, w2v, R2);
  gather_rows_kernel<<<R1, 256, 0, stream>>>(n1i, emb, e1);

  // ---- local aggregator ----
  local_agg_kernel<<<Bb, 256, 0, stream>>>(h0, adj, la_a, hloc);

  // ---- target aggregator -> d_out first half ----
  gemm_gather_store_kernel<<<Bb / 16, 128, 0, stream>>>(targets, emb, taw2f, tw2b);
  gemm_gate_kernel<<<BL / 16, 128, 0, stream>>>(h0, taw1f, tw2b, ta_b, mask, dout);

  const float* w1last0 = gw1 + 256 * 256;
  const float* w1last1 = gw1 + 257 * 256 + 256 * 256;

  // ---- global aggregator: n_hop 0, hp 0 (S=20) ----
  gemm_alpha_kernel<<<R1 / 16, 128, 0, stream>>>(sess, e1, nullptr, emb, w1v, gw1f0,
                                                 w1last0, gw2, 240, alphab);
  softmax_aggn_kernel<<<BL, 256, 0, stream>>>(alphab, e1, nullptr, emb, aggb);
  gemm_relu_cat_kernel<<<BL / 16, 128, 0, stream>>>(h0, aggb, gw3f0, ent0n);

  // ---- n_hop 0, hp 1 (S=240) ----
  gemm_alpha_kernel<<<R2 / 16, 128, 0, stream>>>(sess, nullptr, n2i, emb, w2v, gw1f0,
                                                 w1last0, gw2, 2880, alphab);
  softmax_aggn_kernel<<<R1, 256, 0, stream>>>(alphab, nullptr, n2i, emb, aggb);
  gemm_relu_cat_kernel<<<R1 / 16, 128, 0, stream>>>(e1, aggb, gw3f0, ent1n);

  // ---- n_hop 1, hp 0 (S=20) ----
  gemm_alpha_kernel<<<R1 / 16, 128, 0, stream>>>(sess, ent1n, nullptr, emb, w1v, gw1f1,
                                                 w1last1, gw2 + 256, 240, alphab);
  softmax_aggn_kernel<<<BL, 256, 0, stream>>>(alphab, ent1n, nullptr, emb, aggb);
  gemm_relu_cat_kernel<<<BL / 16, 128, 0, stream>>>(ent0n, aggb, gw3f1, hglob);

  // ---- outputs ----
  addout_kernel<<<BL, 256, 0, stream>>>(hloc, hglob, dout);
  con_part_kernel<<<Bb, 256, 0, stream>>>(hloc, hglob, conp);
  con_reduce_kernel<<<1, 128, 0, stream>>>(conp, dout, out_size - 1);
}
